// TransformerBlock_19189913879158
// MI455X (gfx1250) — matrix operand order repaired and checked by reading, not compiled, not run on hardware
//
#include <hip/hip_runtime.h>
#include <hip/hip_bf16.h>

// ---------------------------------------------------------------------------
// Transformer block (attention + SwiGLU FFN) for MI455X / gfx1250.
// All GEMMs + attention matmuls via v_wmma_f32_16x16x32_bf16 (wave32).
// fp32 inputs cast to bf16 once per launch; accumulation stays fp32.
// GEMM wave tile 64x64 (16 WMMA : 16 b128 loads per K-step) for L2-BW headroom.
// Attention K-chunk staged to LDS via GLOBAL_LOAD_ASYNC_TO_LDS_B128 (ASYNCcnt
// path) using the builtin's probed signature: (v4i AS1*, v4i AS3*, imm, imm).
// ---------------------------------------------------------------------------

typedef __bf16 bf16;
typedef __attribute__((ext_vector_type(16))) __bf16 v16bf;
typedef __attribute__((ext_vector_type(8)))  __bf16 v8bf;
typedef __attribute__((ext_vector_type(8)))  float  v8f;
typedef int v4i __attribute__((vector_size(16)));
typedef __attribute__((address_space(1))) v4i* gv4i_ptr;
typedef __attribute__((address_space(3))) v4i* lv4i_ptr;

#define D_MODEL 2048
#define N_HEADS 16
#define HEAD_DIM 128
#define SEQ 2048
#define BATCH 2
#define TOKENS (BATCH * SEQ)          // 4096
#define HIDDEN 5461
#define HPAD 5632                     // 256 * 22, zero-padded

#if defined(__gfx1250__) && __has_builtin(__builtin_amdgcn_global_load_async_to_lds_b128)
#define ATTN_ASYNC 1
#endif

#define WMMA_BF16(a, b, c) \
  __builtin_amdgcn_wmma_f32_16x16x32_bf16(false, (a), false, (b), (short)0, (c), false, false)

// ---- WMMA fragment loaders (contiguous K order k=16*half+s, same for A and B) ------------------
// A (16x32 bf16): lane<16 -> row m=lane, K in [0..15]; lane>=16 -> K [16..31]
__device__ __forceinline__ v16bf load_frag_a(const bf16* A, int lda, int m0, int kb, int lane) {
  int m = m0 + (lane & 15);
  int half = lane >> 4;
  const bf16* p = A + (size_t)m * lda + kb + 16 * half;
  union { v16bf v; v8bf h[2]; } u;
  u.h[0] = *(const v8bf*)(p);
  u.h[1] = *(const v8bf*)(p + 8);
  return u.v;
}
// B (32x16 bf16), W stored [N,K] row-major: lane%16 = n, lane/16 = K-half, 16 contiguous K elems
__device__ __forceinline__ v16bf load_frag_b(const bf16* W, int ldw, int n0, int kb, int lane) {
  int n = n0 + (lane & 15);
  int half = lane >> 4;
  return *(const v16bf*)(W + (size_t)n * ldw + kb + 16 * half);
}

// ---------------------------------------------------------------------------
// fp32 -> bf16 cast with zero padding: src [R,C] -> dst [Rp,Cp]
__global__ void cast_pad_kernel(const float* __restrict__ src, bf16* __restrict__ dst,
                                int R, int C, int Rp, int Cp) {
  size_t idx = (size_t)blockIdx.x * blockDim.x + threadIdx.x;
  size_t total = (size_t)Rp * Cp;
  if (idx >= total) return;
  int r = (int)(idx / Cp);
  int c = (int)(idx % Cp);
  float v = (r < R && c < C) ? src[(size_t)r * C + c] : 0.0f;
  dst[idx] = (bf16)v;
}

// ---------------------------------------------------------------------------
// RMSNorm: one block (256 thr) per token row; fp32 in, bf16 out
__global__ __launch_bounds__(256) void rmsnorm_kernel(const float* __restrict__ X,
                                                      const float* __restrict__ g,
                                                      bf16* __restrict__ Out) {
  __shared__ float red[8];
  int t = blockIdx.x;
  const float* x = X + (size_t)t * D_MODEL;
  float ss = 0.0f;
  for (int c = threadIdx.x; c < D_MODEL; c += 256) {
    float v = x[c];
    ss += v * v;
  }
#pragma unroll
  for (int off = 16; off >= 1; off >>= 1) ss += __shfl_xor(ss, off, 32);
  if ((threadIdx.x & 31) == 0) red[threadIdx.x >> 5] = ss;
  __syncthreads();
  float tot = 0.0f;
#pragma unroll
  for (int w = 0; w < 8; ++w) tot += red[w];
  float inv = rsqrtf(tot * (1.0f / D_MODEL) + 1e-6f);
  bf16* o = Out + (size_t)t * D_MODEL;
  for (int c = threadIdx.x; c < D_MODEL; c += 256) o[c] = (bf16)(x[c] * inv * g[c]);
}

// ---------------------------------------------------------------------------
// RoPE applied in place to Q and K (bf16 [T, H*HD]); freq fp32 [S, HD/2, 2, 2]
__global__ __launch_bounds__(64) void rope_kernel(bf16* __restrict__ Q, bf16* __restrict__ K,
                                                  const float* __restrict__ freq) {
  int th = blockIdx.x;               // 0 .. T*H-1
  int t = th >> 4;                   // token
  int h = th & 15;                   // head
  int s = t & (SEQ - 1);             // position within sequence
  int i = threadIdx.x;               // 0..63 rotation pair
  const float* f = freq + (((size_t)s * (HEAD_DIM / 2) + i) * 4);
  size_t base = (size_t)t * D_MODEL + h * HEAD_DIM + 2 * i;
  float f00 = f[0], f01 = f[1], f10 = f[2], f11 = f[3];
  float q0 = (float)Q[base], q1 = (float)Q[base + 1];
  float k0 = (float)K[base], k1 = (float)K[base + 1];
  Q[base]     = (bf16)(q0 * f00 + q1 * f01);
  Q[base + 1] = (bf16)(q0 * f10 + q1 * f11);
  K[base]     = (bf16)(k0 * f00 + k1 * f01);
  K[base + 1] = (bf16)(k0 * f10 + k1 * f11);
}

// ---------------------------------------------------------------------------
// GEMM core: one wave computes a 64(M) x 64(N) tile, K-step 32.
// 16 WMMAs : 16 b128 loads per step -> 1.0 loads/WMMA.
__device__ __forceinline__ void wave_tile_gemm64(const bf16* __restrict__ A,
                                                 const bf16* __restrict__ W,
                                                 int K, int m0, int n0, int lane,
                                                 v8f acc[4][4]) {
  for (int kb = 0; kb < K; kb += 32) {
    __builtin_prefetch(A + (size_t)(m0 + (lane & 15)) * K + kb + 64, 0, 1);
    v16bf a[4];
#pragma unroll
    for (int s = 0; s < 4; ++s) a[s] = load_frag_a(A, K, m0 + 16 * s, kb, lane);
#pragma unroll
    for (int j = 0; j < 4; ++j) {
      v16bf b = load_frag_b(W, K, n0 + 16 * j, kb, lane);
#pragma unroll
      for (int s = 0; s < 4; ++s) acc[s][j] = WMMA_BF16(a[s], b, acc[s][j]);
    }
  }
}

// C = A * W^T -> bf16.  grid: (N/256, M/128), block 256 (8 waves, 2x4 tiling)
__global__ __launch_bounds__(256) void gemm_bf16_kernel(const bf16* __restrict__ A,
                                                        const bf16* __restrict__ W,
                                                        bf16* __restrict__ C,
                                                        int N, int K) {
  int lane = threadIdx.x & 31;
  int wave = threadIdx.x >> 5;
  int m0 = blockIdx.y * 128 + (wave >> 2) * 64;
  int n0 = blockIdx.x * 256 + (wave & 3) * 64;
  v8f acc[4][4] = {};
  wave_tile_gemm64(A, W, K, m0, n0, lane, acc);
  int half = lane >> 4, nl = lane & 15;
#pragma unroll
  for (int s = 0; s < 4; ++s)
#pragma unroll
    for (int j = 0; j < 4; ++j)
#pragma unroll
      for (int i = 0; i < 8; ++i) {
        int row = m0 + 16 * s + i + 8 * half;
        int col = n0 + 16 * j + nl;
        C[(size_t)row * N + col] = (bf16)acc[s][j][i];
      }
}

// C = A * W^T + bias + res -> fp32 (bias may be null)
__global__ __launch_bounds__(256) void gemm_f32_res_kernel(const bf16* __restrict__ A,
                                                           const bf16* __restrict__ W,
                                                           const float* __restrict__ bias,
                                                           const float* __restrict__ res,
                                                           float* __restrict__ C,
                                                           int N, int K) {
  int lane = threadIdx.x & 31;
  int wave = threadIdx.x >> 5;
  int m0 = blockIdx.y * 128 + (wave >> 2) * 64;
  int n0 = blockIdx.x * 256 + (wave & 3) * 64;
  v8f acc[4][4] = {};
  wave_tile_gemm64(A, W, K, m0, n0, lane, acc);
  int half = lane >> 4, nl = lane & 15;
#pragma unroll
  for (int s = 0; s < 4; ++s)
#pragma unroll
    for (int j = 0; j < 4; ++j)
#pragma unroll
      for (int i = 0; i < 8; ++i) {
        int row = m0 + 16 * s + i + 8 * half;
        int col = n0 + 16 * j + nl;
        float v = acc[s][j][i] + res[(size_t)row * N + col];
        if (bias) v += bias[col];
        C[(size_t)row * N + col] = v;
      }
}

// Fused SwiGLU: act = silu(A*W1^T) * (A*W3^T) -> bf16.
// One wave computes 32(M) x 64(N) for both weight mats (shared A-frags).
// grid (HPAD/256, M/64), block 256 (2x4 waves)
__global__ __launch_bounds__(256) void gemm_swiglu_kernel(const bf16* __restrict__ A,
                                                          const bf16* __restrict__ W1,
                                                          const bf16* __restrict__ W3,
                                                          bf16* __restrict__ Act,
                                                          int N, int K) {
  int lane = threadIdx.x & 31;
  int wave = threadIdx.x >> 5;
  int m0 = blockIdx.y * 64 + (wave >> 2) * 32;
  int n0 = blockIdx.x * 256 + (wave & 3) * 64;
  v8f acc1[2][4] = {};
  v8f acc3[2][4] = {};
  for (int kb = 0; kb < K; kb += 32) {
    v16bf a0 = load_frag_a(A, K, m0, kb, lane);
    v16bf a1 = load_frag_a(A, K, m0 + 16, kb, lane);
#pragma unroll
    for (int j = 0; j < 4; ++j) {
      v16bf b1 = load_frag_b(W1, K, n0 + 16 * j, kb, lane);
      v16bf b3 = load_frag_b(W3, K, n0 + 16 * j, kb, lane);
      acc1[0][j] = WMMA_BF16(a0, b1, acc1[0][j]);
      acc1[1][j] = WMMA_BF16(a1, b1, acc1[1][j]);
      acc3[0][j] = WMMA_BF16(a0, b3, acc3[0][j]);
      acc3[1][j] = WMMA_BF16(a1, b3, acc3[1][j]);
    }
  }
  int half = lane >> 4, nl = lane & 15;
#pragma unroll
  for (int s = 0; s < 2; ++s)
#pragma unroll
    for (int j = 0; j < 4; ++j)
#pragma unroll
      for (int i = 0; i < 8; ++i) {
        int row = m0 + 16 * s + i + 8 * half;
        int col = n0 + 16 * j + nl;
        float x1 = acc1[s][j][i];
        float x3 = acc3[s][j][i];
        float silu = x1 / (1.0f + __expf(-x1));
        Act[(size_t)row * N + col] = (bf16)(silu * x3);
      }
}

// ---------------------------------------------------------------------------
// Flash attention (no causal mask). One wave per (b, h, 16-query block).
// Online softmax; K rows and V^T staged in LDS; all matmuls via WMMA.
__global__ __launch_bounds__(32) void attn_kernel(const bf16* __restrict__ Q,
                                                  const bf16* __restrict__ Kx,
                                                  const bf16* __restrict__ V,
                                                  bf16* __restrict__ O) {
  __shared__ __align__(32) bf16 lds_k[32 * HEAD_DIM];   // K chunk, row-major  (8 KB)
  __shared__ __align__(32) bf16 lds_vt[HEAD_DIM * 32];  // V chunk, transposed (8 KB)
  __shared__ __align__(32) bf16 lds_p[16 * 32];         // P tile transpose    (1 KB)

  const int lane = threadIdx.x;
  const int half = lane >> 4;
  const int nl = lane & 15;
  const int qb = blockIdx.x * 16;
  const int h = blockIdx.y;
  const int b = blockIdx.z;
  const float scale = 0.08838834764831845f;  // 1/sqrt(128)

  const size_t bbase = (size_t)b * SEQ * D_MODEL + (size_t)h * HEAD_DIM;
  const bf16* Qp = Q + bbase;
  const bf16* Kp = Kx + bbase;
  const bf16* Vp = V + bbase;

  // Q tile as 4 A-fragments (K-steps of 32 over head_dim)
  v16bf qf[4];
#pragma unroll
  for (int s = 0; s < 4; ++s) {
    const bf16* p = Qp + (size_t)(qb + nl) * D_MODEL + 32 * s + 16 * half;
    union { v16bf v; v8bf h2[2]; } u;
    u.h2[0] = *(const v8bf*)(p);
    u.h2[1] = *(const v8bf*)(p + 8);
    qf[s] = u.v;
  }

  v8f o_acc[8] = {};
  float rmax[8], rsum[8];
#pragma unroll
  for (int i = 0; i < 8; ++i) { rmax[i] = -3.0e38f; rsum[i] = 0.0f; }

  for (int kc = 0; kc < SEQ; kc += 32) {
#ifdef ATTN_ASYNC
    // K chunk via async DMA to LDS (ASYNCcnt path); lane r stages key row kc+r
#pragma unroll
    for (int t = 0; t < 16; ++t) {
      const bf16* g = Kp + (size_t)(kc + lane) * D_MODEL + 8 * t;
      __builtin_amdgcn_global_load_async_to_lds_b128(
          (gv4i_ptr)(g),
          (lv4i_ptr)(lds_k + lane * HEAD_DIM + 8 * t),
          0, 0);
    }
#else
#pragma unroll
    for (int t = 0; t < 16; ++t) {
      v8bf kk = *(const v8bf*)(Kp + (size_t)(kc + lane) * D_MODEL + 8 * t);
      *(v8bf*)(lds_k + lane * HEAD_DIM + 8 * t) = kk;
    }
#endif
    // V chunk transposed on the fly (needs element shuffle; stays on VALU path)
#pragma unroll
    for (int t = 0; t < 16; ++t) {
      v8bf vv = *(const v8bf*)(Vp + (size_t)(kc + lane) * D_MODEL + 8 * t);
#pragma unroll
      for (int e = 0; e < 8; ++e) lds_vt[(8 * t + e) * 32 + lane] = vv[e];
    }
#ifdef ATTN_ASYNC
#if __has_builtin(__builtin_amdgcn_s_wait_asynccnt)
    __builtin_amdgcn_s_wait_asynccnt(0);
#else
    asm volatile("s_wait_asynccnt 0x0" ::: "memory");
#endif
#endif
    __syncthreads();

    // scores: S[16q x 32k] = Q(16x128) * Kchunk^T
    v8f c0 = {}, c1 = {};
#pragma unroll
    for (int s = 0; s < 4; ++s) {
      v16bf bk0 = *(const v16bf*)(lds_k + (0 + nl) * HEAD_DIM + 32 * s + 16 * half);
      v16bf bk1 = *(const v16bf*)(lds_k + (16 + nl) * HEAD_DIM + 32 * s + 16 * half);
      c0 = WMMA_BF16(qf[s], bk0, c0);
      c1 = WMMA_BF16(qf[s], bk1, c1);
    }

    // online softmax update (row i + 8*half, values across 16-lane group)
#pragma unroll
    for (int i = 0; i < 8; ++i) {
      float s0 = c0[i] * scale;
      float s1 = c1[i] * scale;
      float t = fmaxf(s0, s1);
#pragma unroll
      for (int off = 1; off < 16; off <<= 1) t = fmaxf(t, __shfl_xor(t, off, 32));
      float mnew = fmaxf(rmax[i], t);
      float alpha = __expf(rmax[i] - mnew);
      rmax[i] = mnew;
      float p0 = __expf(s0 - mnew);
      float p1 = __expf(s1 - mnew);
      float ps = p0 + p1;
#pragma unroll
      for (int off = 1; off < 16; off <<= 1) ps += __shfl_xor(ps, off, 32);
      rsum[i] = rsum[i] * alpha + ps;
      int row = i + 8 * half;
      lds_p[row * 32 + nl] = (bf16)p0;
      lds_p[row * 32 + 16 + nl] = (bf16)p1;
#pragma unroll
      for (int j = 0; j < 8; ++j) o_acc[j][i] *= alpha;
    }
    __syncthreads();

    // P tile as A-fragment
    union { v16bf v; v8bf h2[2]; } pu;
    pu.h2[0] = *(const v8bf*)(lds_p + nl * 32 + 16 * half);
    pu.h2[1] = *(const v8bf*)(lds_p + nl * 32 + 16 * half + 8);

    // O += P(16x32) * Vchunk(32x128), per 16-wide hd slice
#pragma unroll
    for (int j = 0; j < 8; ++j) {
      v16bf bv = *(const v16bf*)(lds_vt + (16 * j + nl) * 32 + 16 * half);
      o_acc[j] = WMMA_BF16(pu.v, bv, o_acc[j]);
    }
    __syncthreads();
  }

  // epilogue: normalize and store bf16
#pragma unroll
  for (int j = 0; j < 8; ++j)
#pragma unroll
    for (int i = 0; i < 8; ++i) {
      int row = qb + i + 8 * half;
      int hd = 16 * j + nl;
      O[(size_t)(b * SEQ + row) * D_MODEL + h * HEAD_DIM + hd] = (bf16)(o_acc[j][i] / rsum[i]);
    }
}

// ---------------------------------------------------------------------------
extern "C" void kernel_launch(void* const* d_in, const int* in_sizes, int n_in,
                              void* d_out, int out_size, void* d_ws, size_t ws_size,
                              hipStream_t stream) {
  (void)in_sizes; (void)n_in; (void)out_size; (void)ws_size;
  const float* x      = (const float*)d_in[0];
  const float* freq   = (const float*)d_in[1];
  const float* wq     = (const float*)d_in[2];
  const float* wk     = (const float*)d_in[3];
  const float* wv     = (const float*)d_in[4];
  const float* wo     = (const float*)d_in[5];
  const float* bo     = (const float*)d_in[6];
  const float* w1     = (const float*)d_in[7];
  const float* w3     = (const float*)d_in[8];
  const float* w2     = (const float*)d_in[9];
  const float* g_attn = (const float*)d_in[10];
  const float* g_ffn  = (const float*)d_in[11];
  float* out = (float*)d_out;

  // --- workspace layout (all offsets 256B aligned; ~221 MB total) ---
  char* ws = (char*)d_ws;
  size_t off = 0;
  auto alloc = [&](size_t bytes) -> char* {
    char* p = ws + off;
    off += (bytes + 255) & ~(size_t)255;
    return p;
  };
  const size_t DD   = (size_t)D_MODEL * D_MODEL;      // 4.19M
  const size_t TD   = (size_t)TOKENS * D_MODEL;       // 8.39M
  const size_t HD2  = (size_t)HPAD * D_MODEL;         // 11.53M
  bf16* wqb = (bf16*)alloc(DD * 2);
  bf16* wkb = (bf16*)alloc(DD * 2);
  bf16* wvb = (bf16*)alloc(DD * 2);
  bf16* wob = (bf16*)alloc(DD * 2);
  bf16* w1b = (bf16*)alloc(HD2 * 2);
  bf16* w3b = (bf16*)alloc(HD2 * 2);
  bf16* w2b = (bf16*)alloc(HD2 * 2);
  bf16* xn  = (bf16*)alloc(TD * 2);                   // reused as hn
  bf16* qb  = (bf16*)alloc(TD * 2);                   // q/k/v region reused for FFN act
  bf16* kb  = (bf16*)alloc(TD * 2);
  bf16* vb  = (bf16*)alloc(TD * 2);
  bf16* ob  = (bf16*)alloc(TD * 2);
  float* hres = (float*)alloc(TD * 4);
  bf16* hn  = xn;                                     // alias: xn dead after QKV
  bf16* act = qb;                                     // alias: q/k/v dead after attention

  auto cgrid = [](size_t n) { return (unsigned)((n + 255) / 256); };

  // 1) weight casts (zero-padded where needed)
  cast_pad_kernel<<<cgrid(DD), 256, 0, stream>>>(wq, wqb, D_MODEL, D_MODEL, D_MODEL, D_MODEL);
  cast_pad_kernel<<<cgrid(DD), 256, 0, stream>>>(wk, wkb, D_MODEL, D_MODEL, D_MODEL, D_MODEL);
  cast_pad_kernel<<<cgrid(DD), 256, 0, stream>>>(wv, wvb, D_MODEL, D_MODEL, D_MODEL, D_MODEL);
  cast_pad_kernel<<<cgrid(DD), 256, 0, stream>>>(wo, wob, D_MODEL, D_MODEL, D_MODEL, D_MODEL);
  cast_pad_kernel<<<cgrid(HD2), 256, 0, stream>>>(w1, w1b, HIDDEN, D_MODEL, HPAD, D_MODEL);
  cast_pad_kernel<<<cgrid(HD2), 256, 0, stream>>>(w3, w3b, HIDDEN, D_MODEL, HPAD, D_MODEL);
  cast_pad_kernel<<<cgrid(HD2), 256, 0, stream>>>(w2, w2b, D_MODEL, HIDDEN, D_MODEL, HPAD);

  // 2) attention sublayer
  rmsnorm_kernel<<<TOKENS, 256, 0, stream>>>(x, g_attn, xn);
  dim3 gproj(D_MODEL / 256, TOKENS / 128);
  gemm_bf16_kernel<<<gproj, 256, 0, stream>>>(xn, wqb, qb, D_MODEL, D_MODEL);
  gemm_bf16_kernel<<<gproj, 256, 0, stream>>>(xn, wkb, kb, D_MODEL, D_MODEL);
  gemm_bf16_kernel<<<gproj, 256, 0, stream>>>(xn, wvb, vb, D_MODEL, D_MODEL);
  rope_kernel<<<TOKENS * N_HEADS, 64, 0, stream>>>(qb, kb, freq);
  attn_kernel<<<dim3(SEQ / 16, N_HEADS, BATCH), 32, 0, stream>>>(qb, kb, vb, ob);
  gemm_f32_res_kernel<<<gproj, 256, 0, stream>>>(ob, wob, bo, x, hres, D_MODEL, D_MODEL);

  // 3) SwiGLU FFN sublayer
  rmsnorm_kernel<<<TOKENS, 256, 0, stream>>>(hres, g_ffn, hn);
  gemm_swiglu_kernel<<<dim3(HPAD / 256, TOKENS / 64), 256, 0, stream>>>(hn, w1b, w3b, act, HPAD, D_MODEL);
  gemm_f32_res_kernel<<<gproj, 256, 0, stream>>>(act, w2b, nullptr, hres, out, D_MODEL, HPAD);
}
